// HitTagNetTwo_76708115906682
// MI455X (gfx1250) — matrix operand order repaired and checked by reading, not compiled, not run on hardware
//
#include <hip/hip_runtime.h>
#include <hip/hip_bf16.h>
#include <math.h>

typedef __attribute__((ext_vector_type(16))) _Float16 v16h;
typedef __attribute__((ext_vector_type(8)))  float    v8f;
typedef __attribute__((ext_vector_type(4)))  float    v4f;

#define D_IN 32
#define H_HID 128

__global__ __launch_bounds__(256, 1)
void hittag_wmma_kernel(const float* __restrict__ x,    // N x 32
                        const float* __restrict__ WA1,  // 32 x 128
                        const float* __restrict__ bA1,  // 128
                        const float* __restrict__ WA2,  // 128 x 2
                        const float* __restrict__ bA2,  // 2
                        const float* __restrict__ WB1,
                        const float* __restrict__ bB1,
                        const float* __restrict__ WB2,
                        const float* __restrict__ bB2,
                        float* __restrict__ out,        // N x 3
                        int N)
{
    const int lane         = threadIdx.x & 31;
    const int waveInBlock  = threadIdx.x >> 5;
    const int wavesPerBlk  = blockDim.x >> 5;
    const int globalWave   = blockIdx.x * wavesPerBlk + waveInBlock;
    const int totalWaves   = gridDim.x * wavesPerBlk;

    const int n  = lane & 15;   // column within 16-wide tile (B/C/D lane index)
    const int hi = lane >> 4;   // half-wave selector

    // 16 rows x (pA,pB) staging per wave: [row][0]=pA, [row][1]=pB
    __shared__ float ldsbuf[8][16][2];
    volatile float* vl = &ldsbuf[waveInBlock][0][0];

    // ---------------- loop-invariant: weights -> f16 fragments in registers ----
    // B-fragment layout (16-bit B, 32x16): matches A: element j <-> K = 8*hi + j (j<8),
    // K = 16 + 8*hi + (j-8) (j>=8).
    v16h  BA[8], BB[8];
    float b1A[8], b1B[8], wdA[8], wdB[8];
#pragma unroll
    for (int nt = 0; nt < 8; ++nt) {
        const int h = nt * 16 + n;
#pragma unroll
        for (int j = 0; j < 16; ++j) {
            const int k = (j < 8) ? (8 * hi + j) : (16 + 8 * hi + (j - 8));
            BA[nt][j] = (_Float16)WA1[k * H_HID + h];
            BB[nt][j] = (_Float16)WB1[k * H_HID + h];
        }
        b1A[nt] = bA1[h];
        b1B[nt] = bB1[h];
        wdA[nt] = WA2[h * 2 + 0] - WA2[h * 2 + 1];
        wdB[nt] = WB2[h * 2 + 0] - WB2[h * 2 + 1];
    }
    const float bdA = bA2[0] - bA2[1];
    const float bdB = bB2[0] - bB2[1];
    const v8f zero_c = {};

    const int numTiles = (N + 15) >> 4;
    for (int t = globalWave; t < numTiles; t += totalWaves) {
        const int base = t << 4;
        int r = base + n;               // A-matrix: lane%16 = row
        if (r >= N) r = N - 1;          // clamp so WMMA runs with EXEC all-1s

        // A layout (16-bit A, 16x32): lanes 0-15: V0-3 = K0..7, V4-7 = K16..23
        //                             lanes16-31: V0-3 = K8..15, V4-7 = K24..31
        const float* xr = x + (size_t)r * D_IN + hi * 8;
        const v4f f0 = *(const v4f*)(xr + 0);
        const v4f f1 = *(const v4f*)(xr + 4);
        const v4f g0 = *(const v4f*)(xr + 16);
        const v4f g1 = *(const v4f*)(xr + 20);

        // prefetch next tile's row chunks for this lane (grid-stride reuse)
        {
            int tn = t + totalWaves;
            if (tn < numTiles) {
                int rn = (tn << 4) + n;
                if (rn >= N) rn = N - 1;
                const float* xp = x + (size_t)rn * D_IN + hi * 8;
                __builtin_prefetch(xp, 0, 1);
                __builtin_prefetch(xp + 16, 0, 1);
            }
        }

        v16h a;
#pragma unroll
        for (int j = 0; j < 4; ++j) {
            a[j]      = (_Float16)f0[j];
            a[4 + j]  = (_Float16)f1[j];
            a[8 + j]  = (_Float16)g0[j];
            a[12 + j] = (_Float16)g1[j];
        }

        float pA[8], pB[8];
#pragma unroll
        for (int v = 0; v < 8; ++v) { pA[v] = 0.f; pB[v] = 0.f; }

#pragma unroll
        for (int nt = 0; nt < 8; ++nt) {
            // Issue both nets' WMMAs back-to-back: independent TRANS-class work
            // lets the scheduler cover the WMMA->VALU RAW hazard without v_nops.
            v8f cA = __builtin_amdgcn_wmma_f32_16x16x32_f16(
                false, a, false, BA[nt], (short)0, zero_c, false, false);
            v8f cB = __builtin_amdgcn_wmma_f32_16x16x32_f16(
                false, a, false, BB[nt], (short)0, zero_c, false, false);
#pragma unroll
            for (int v = 0; v < 8; ++v) {
                float uA = fmaxf(cA[v] + b1A[nt], 0.f);    // bias + ReLU
                pA[v] = fmaf(uA, wdA[nt], pA[v]);          // layer-2 logit diff
                float uB = fmaxf(cB[v] + b1B[nt], 0.f);
                pB[v] = fmaf(uB, wdB[nt], pB[v]);
            }
        }

        // Butterfly-reduce over the 16 columns (masks 1,2,4,8 stay in each half)
#pragma unroll
        for (int m = 1; m <= 8; m <<= 1) {
#pragma unroll
            for (int v = 0; v < 8; ++v) {
                pA[v] += __shfl_xor(pA[v], m, 32);
                pB[v] += __shfl_xor(pB[v], m, 32);
            }
        }

        // Stage 16 rows x (pA,pB) through LDS (wave-internal; LDS is in-order
        // within a wave, volatile + wave_barrier stop compiler reordering).
        if (n == 0) {
#pragma unroll
            for (int v = 0; v < 8; ++v) {
                const int row = 8 * hi + v;    // C/D layout: VGPR v -> row v (+8 upper half)
                vl[2 * row + 0] = pA[v];
                vl[2 * row + 1] = pB[v];
            }
        }
        __builtin_amdgcn_wave_barrier();

        // One lane per row computes the sigmoid epilogue once; rows are
        // contiguous -> coalesced 12B (b96) stores.
        const int row = base + n;
        if (hi == 0 && row < N) {
            const float dA  = vl[2 * n + 0] + bdA;         // logitA0 - logitA1
            const float dB  = vl[2 * n + 1] + bdB;
            const float yA0 = 1.f / (1.f + __expf(-dA));
            const float yA1 = 1.f - yA0;
            const float yB0 = 1.f / (1.f + __expf(-dB));
            const float yB1 = 1.f - yB0;
            const bool keep = dA >= 0.f;                   // argmax(yA)==0
            float3 o;
            o.x = keep ? yB0 * yA0 : 0.f;
            o.y = keep ? yB1 * yA0 : yA0;
            o.z = yA1;
            *(float3*)(out + (size_t)row * 3) = o;
        }
        __builtin_amdgcn_wave_barrier();   // keep next iteration's LDS writes ordered
    }
}

extern "C" void kernel_launch(void* const* d_in, const int* in_sizes, int n_in,
                              void* d_out, int out_size, void* d_ws, size_t ws_size,
                              hipStream_t stream)
{
    (void)n_in; (void)out_size; (void)d_ws; (void)ws_size;
    const float* x   = (const float*)d_in[0];
    // d_in[1] = x_coords (int32, unused by the reference computation)
    const float* WA1 = (const float*)d_in[2];
    const float* bA1 = (const float*)d_in[3];
    const float* WA2 = (const float*)d_in[4];
    const float* bA2 = (const float*)d_in[5];
    const float* WB1 = (const float*)d_in[6];
    const float* bB1 = (const float*)d_in[7];
    const float* WB2 = (const float*)d_in[8];
    const float* bB2 = (const float*)d_in[9];
    float* out = (float*)d_out;

    const int N = in_sizes[0] / D_IN;
    const int numTiles = (N + 15) / 16;
    const int wavesPerBlock = 8;              // 256 threads = 8 wave32
    int blocks = (numTiles + wavesPerBlock - 1) / wavesPerBlock;
    if (blocks > 1024) blocks = 1024;
    if (blocks < 1)    blocks = 1;

    hipLaunchKernelGGL(hittag_wmma_kernel, dim3(blocks), dim3(256), 0, stream,
                       x, WA1, bA1, WA2, bA2, WB1, bB1, WB2, bB2, out, N);
}
